// RNN_39118562132539
// MI455X (gfx1250) — matrix operand order repaired and checked by reading, not compiled, not run on hardware
//
#include <hip/hip_runtime.h>

// Problem constants
constexpr int Bz = 32;
constexpr int Tz = 2048;
constexpr int Iz = 128;
constexpr int Hz = 256;
constexpr int Gz = 4 * Hz;   // 1024 gate dims (i,f,g,o)
constexpr int Oz = 64;

typedef __attribute__((ext_vector_type(16))) __bf16 v16bf;
typedef __attribute__((ext_vector_type(8)))  __bf16 v8bf;
typedef __attribute__((ext_vector_type(8)))  float  v8f;

union ABfrag { v16bf v; struct { v8bf lo, hi; } s; };

__device__ __forceinline__ unsigned short f2bf(float f) {
  __bf16 h = (__bf16)f;
  return __builtin_bit_cast(unsigned short, h);
}
__device__ __forceinline__ float bf2f(unsigned short u) {
  __bf16 h = __builtin_bit_cast(__bf16, u);
  return (float)h;
}
__device__ __forceinline__ float sigmoidf_fast(float x) {
  return 1.0f / (1.0f + __expf(-x));
}
__device__ __forceinline__ v8f wmma_bf16(v16bf a, v16bf b, v8f c) {
  // D = A(16x32 bf16) x B(32x16 bf16) + C(16x16 f32)
  return __builtin_amdgcn_wmma_f32_16x16x32_bf16(
      /*neg_a=*/false, a, /*neg_b=*/false, b,
      /*c_mod=*/(short)0, c, /*reuse_a=*/false, /*reuse_b=*/false);
}

// ---------------------------------------------------------------------------
// Precision conversion kernels
// ---------------------------------------------------------------------------
__global__ void cvt_f32_bf16(const float* __restrict__ src,
                             unsigned short* __restrict__ dst, int n) {
  int i = blockIdx.x * blockDim.x + threadIdx.x;
  if (i < n) dst[i] = f2bf(src[i]);
}

__global__ void bias_sum(const float* __restrict__ b_ih,
                         const float* __restrict__ b_hh,
                         float* __restrict__ bias_g) {
  int i = blockIdx.x * blockDim.x + threadIdx.x;
  if (i < Gz) bias_g[i] = b_ih[i] + b_hh[i];
}

// ---------------------------------------------------------------------------
// Kernel 1: x_proj = x @ W_ih^T + (b_ih + b_hh)    (65536 x 1024, K=128)
// Block = 8 waves; block tile 128(M) x 32(N); wave tile 16 x 32 (2 N-tiles).
// ---------------------------------------------------------------------------
__global__ __launch_bounds__(256) void xproj_gemm(
    const unsigned short* __restrict__ xb,   // (B*T, I) bf16 row-major
    const unsigned short* __restrict__ wb,   // (G, I)  bf16 row-major
    const float* __restrict__ bias,          // (G)
    float* __restrict__ xp) {                // (B*T, G) f32
  const int wave = threadIdx.x >> 5;
  const int lane = threadIdx.x & 31;
  const int l15  = lane & 15;
  const int hi   = (lane & 16) ? 1 : 0;
  const int nblk = blockIdx.x & 31;          // 1024/32
  const int mblk = blockIdx.x >> 5;
  const int mrow = mblk * 128 + wave * 16;
  const int ncol = nblk * 32;

  v8f acc0 = {}, acc1 = {};
#pragma unroll
  for (int kt = 0; kt < Iz / 32; ++kt) {
    ABfrag a;
    const unsigned short* ap =
        xb + (size_t)(mrow + l15) * Iz + kt * 32 + hi * 16;
    a.s.lo = *(const v8bf*)ap;
    a.s.hi = *(const v8bf*)(ap + 8);
    v16bf b0 = *(const v16bf*)(wb + (size_t)(ncol      + l15) * Iz + kt * 32 + hi * 16);
    v16bf b1 = *(const v16bf*)(wb + (size_t)(ncol + 16 + l15) * Iz + kt * 32 + hi * 16);
    acc0 = wmma_bf16(a.v, b0, acc0);
    acc1 = wmma_bf16(a.v, b1, acc1);
  }
  const float bn0 = bias[ncol + l15];
  const float bn1 = bias[ncol + 16 + l15];
#pragma unroll
  for (int r = 0; r < 8; ++r) {
    const int m = mrow + r + hi * 8;              // C layout: M = r + 8*(lane>=16)
    xp[(size_t)m * Gz + ncol + l15]      = acc0[r] + bn0;
    xp[(size_t)m * Gz + ncol + 16 + l15] = acc1[r] + bn1;
  }
}

// ---------------------------------------------------------------------------
// Kernel 2: persistent LSTM scan. Batch split: block b handles rows [16b,16b+16).
// 32 waves; wave w owns gate columns [32w, 32w+32). W_hh fragments live in
// VGPRs across all 2048 steps (2 x 8 x v16bf = 128 VGPRs/wave).
// x_proj slices are double-buffered in LDS via GLOBAL_LOAD_ASYNC_TO_LDS_B128
// (ASYNCcnt), issued one step ahead so the serial loop never waits on HBM.
// ---------------------------------------------------------------------------
__global__ __launch_bounds__(1024) void lstm_scan(
    const unsigned short* __restrict__ whh,  // (G, H) bf16 row-major
    const float* __restrict__ xp,            // (B*T, G) f32
    unsigned short* __restrict__ hs) {       // (B*T, H) bf16
  __shared__ float          sh_xp[2][16 * Gz]; // 128 KB (x_proj double buffer)
  __shared__ unsigned short sh_h[16 * Hz];     //   8 KB (bf16 h, A-matrix)
  __shared__ unsigned short sh_g[16 * Gz];     //  32 KB (bf16 recurrent gates)
  __shared__ float          sh_c[16 * Hz];     //  16 KB (f32 cell state)

  const int tid  = threadIdx.x;
  const int wave = tid >> 5;
  const int lane = tid & 31;
  const int l15  = lane & 15;
  const int hi   = (lane & 16) ? 1 : 0;
  const int mbase = blockIdx.x * 16;

  for (int e = tid; e < 16 * Hz; e += 1024) { sh_h[e] = 0; sh_c[e] = 0.0f; }

  // Register-resident W_hh^T fragments. B[k][n] = W_hh[n][k]; per-lane:
  // n = base + (lane&15), k = kt*32 + 16*(lane>=16) .. +15 (contiguous row).
  v16bf Bf[2][8];
#pragma unroll
  for (int nt = 0; nt < 2; ++nt) {
#pragma unroll
    for (int kt = 0; kt < 8; ++kt) {
      const int n = wave * 32 + nt * 16 + l15;
      Bf[nt][kt] = *(const v16bf*)(whh + (size_t)n * Hz + kt * 32 + hi * 16);
    }
  }

  const float* xp_blk = xp + (size_t)mbase * Tz * Gz;

  // Prologue: async-stage the t=0 x_proj slice (16 rows x 1024 f32) into buf 0.
  // Chunk c = i*1024+tid covers 4 consecutive f32 (16B): row m=c>>8, col=(c&255)*4.
#pragma unroll
  for (int i = 0; i < 4; ++i) {
    const int c   = i * 1024 + tid;
    const int m   = c >> 8;
    const int col = (c & 255) * 4;
    const float* g = xp_blk + ((size_t)m * Tz + 0) * Gz + col;
    unsigned ldsoff = (unsigned)(uintptr_t)&sh_xp[0][m * Gz + col];
    asm volatile("global_load_async_to_lds_b128 %0, %1, off"
                 :: "v"(ldsoff), "v"((unsigned long long)(uintptr_t)g)
                 : "memory");
  }
  __syncthreads();

  for (int t = 0; t < Tz; ++t) {
    // ---- kick async staging of step t+1 into the other buffer ----
    if (t + 1 < Tz) {
#pragma unroll
      for (int i = 0; i < 4; ++i) {
        const int c   = i * 1024 + tid;
        const int m   = c >> 8;
        const int col = (c & 255) * 4;
        const float* g = xp_blk + ((size_t)m * Tz + (t + 1)) * Gz + col;
        unsigned ldsoff = (unsigned)(uintptr_t)&sh_xp[(t + 1) & 1][m * Gz + col];
        asm volatile("global_load_async_to_lds_b128 %0, %1, off"
                     :: "v"(ldsoff), "v"((unsigned long long)(uintptr_t)g)
                     : "memory");
      }
    }

    // ---- recurrent matmul: (16 x 256) @ (256 x 1024 slice) ----
    v8f acc0 = {}, acc1 = {};
#pragma unroll
    for (int kt = 0; kt < 8; ++kt) {
      ABfrag a;
      const unsigned short* ap = sh_h + l15 * Hz + kt * 32 + hi * 16;
      a.s.lo = *(const v8bf*)ap;
      a.s.hi = *(const v8bf*)(ap + 8);
      acc0 = wmma_bf16(a.v, Bf[0][kt], acc0);
      acc1 = wmma_bf16(a.v, Bf[1][kt], acc1);
    }
#pragma unroll
    for (int r = 0; r < 8; ++r) {
      const int m = r + hi * 8;
      sh_g[m * Gz + wave * 32 + l15]      = f2bf(acc0[r]);
      sh_g[m * Gz + wave * 32 + 16 + l15] = f2bf(acc1[r]);
    }

    // Async loads complete in order: allowing the last 4 (step t+1 batch) to
    // remain outstanding guarantees step t's slice has landed in LDS.
    if (t + 1 < Tz) {
      asm volatile("s_wait_asynccnt 0x4" ::: "memory");
    } else {
      asm volatile("s_wait_asynccnt 0x0" ::: "memory");
    }
    __syncthreads();

    // ---- elementwise gates / state update (16*256 elems, 4 per thread) ----
    const float* xbuf = &sh_xp[t & 1][0];
#pragma unroll
    for (int e4 = 0; e4 < 4; ++e4) {
      const int e = e4 * 1024 + tid;
      const int m = e >> 8;
      const int j = e & 255;
      const float* xrow = xbuf + m * Gz;
      const float gi = bf2f(sh_g[m * Gz + j])          + xrow[j];
      const float gf = bf2f(sh_g[m * Gz + Hz + j])     + xrow[Hz + j];
      const float gg = bf2f(sh_g[m * Gz + 2 * Hz + j]) + xrow[2 * Hz + j];
      const float go = bf2f(sh_g[m * Gz + 3 * Hz + j]) + xrow[3 * Hz + j];
      const float c = sigmoidf_fast(gf) * sh_c[m * Hz + j] +
                      sigmoidf_fast(gi) * tanhf(gg);
      sh_c[m * Hz + j] = c;
      const float h = sigmoidf_fast(go) * tanhf(c);
      const unsigned short hb = f2bf(h);
      sh_h[m * Hz + j] = hb;
      hs[((size_t)(mbase + m) * Tz + t) * Hz + j] = hb;
    }
    __syncthreads();
  }
}

// ---------------------------------------------------------------------------
// Kernel 3: logits = hs @ W_fc^T + b_fc ; softmax over O=64.
// Block = 8 waves; wave tile 16(M) x 64(N) (4 N-tiles), K loop 8.
// Softmax reduction over N done with shfl_xor within 16-lane groups.
// ---------------------------------------------------------------------------
__global__ __launch_bounds__(256) void fc_softmax(
    const unsigned short* __restrict__ hs,   // (B*T, H) bf16
    const unsigned short* __restrict__ wfc,  // (O, H) bf16
    const float* __restrict__ bfc,           // (O)
    float* __restrict__ out) {               // (B*T, O) f32
  const int wave = threadIdx.x >> 5;
  const int lane = threadIdx.x & 31;
  const int l15  = lane & 15;
  const int hi   = (lane & 16) ? 1 : 0;
  const int mrow = blockIdx.x * 128 + wave * 16;

  v8f acc[4] = {};
#pragma unroll
  for (int kt = 0; kt < Hz / 32; ++kt) {
    ABfrag a;
    const unsigned short* ap =
        hs + (size_t)(mrow + l15) * Hz + kt * 32 + hi * 16;
    a.s.lo = *(const v8bf*)ap;
    a.s.hi = *(const v8bf*)(ap + 8);
#pragma unroll
    for (int nt = 0; nt < 4; ++nt) {
      v16bf b = *(const v16bf*)(wfc + (size_t)(nt * 16 + l15) * Hz + kt * 32 + hi * 16);
      acc[nt] = wmma_bf16(a.v, b, acc[nt]);
    }
  }
  float bb[4];
#pragma unroll
  for (int nt = 0; nt < 4; ++nt) bb[nt] = bfc[nt * 16 + l15];

#pragma unroll
  for (int r = 0; r < 8; ++r) {
    float v0 = acc[0][r] + bb[0];
    float v1 = acc[1][r] + bb[1];
    float v2 = acc[2][r] + bb[2];
    float v3 = acc[3][r] + bb[3];
    float mx = fmaxf(fmaxf(v0, v1), fmaxf(v2, v3));
#pragma unroll
    for (int off = 1; off < 16; off <<= 1)
      mx = fmaxf(mx, __shfl_xor(mx, off, 32));
    const float e0 = __expf(v0 - mx);
    const float e1 = __expf(v1 - mx);
    const float e2 = __expf(v2 - mx);
    const float e3 = __expf(v3 - mx);
    float s = e0 + e1 + e2 + e3;
#pragma unroll
    for (int off = 1; off < 16; off <<= 1)
      s += __shfl_xor(s, off, 32);
    const float inv = __builtin_amdgcn_rcpf(s);
    const size_t row = (size_t)(mrow + r + hi * 8);
    out[row * Oz + l15]      = e0 * inv;
    out[row * Oz + 16 + l15] = e1 * inv;
    out[row * Oz + 32 + l15] = e2 * inv;
    out[row * Oz + 48 + l15] = e3 * inv;
  }
}

// ---------------------------------------------------------------------------
extern "C" void kernel_launch(void* const* d_in, const int* in_sizes, int n_in,
                              void* d_out, int out_size, void* d_ws, size_t ws_size,
                              hipStream_t stream) {
  (void)in_sizes; (void)n_in; (void)out_size; (void)ws_size;
  const float* x    = (const float*)d_in[0];
  const float* W_ih = (const float*)d_in[1];
  const float* W_hh = (const float*)d_in[2];
  const float* b_ih = (const float*)d_in[3];
  const float* b_hh = (const float*)d_in[4];
  const float* W_fc = (const float*)d_in[5];
  const float* b_fc = (const float*)d_in[6];
  float* out = (float*)d_out;

  char* ws = (char*)d_ws;
  size_t off = 0;
  auto alloc = [&](size_t bytes) -> void* {
    void* p = ws + off;
    off += (bytes + 255) & ~(size_t)255;
    return p;
  };
  unsigned short* xb   = (unsigned short*)alloc((size_t)Bz * Tz * Iz * 2);
  unsigned short* wihb = (unsigned short*)alloc((size_t)Gz * Iz * 2);
  unsigned short* whhb = (unsigned short*)alloc((size_t)Gz * Hz * 2);
  unsigned short* wfcb = (unsigned short*)alloc((size_t)Oz * Hz * 2);
  float*          biasg = (float*)alloc((size_t)Gz * 4);
  float*          xp    = (float*)alloc((size_t)Bz * Tz * Gz * 4);
  unsigned short* hsb   = (unsigned short*)alloc((size_t)Bz * Tz * Hz * 2);

  const int nx = Bz * Tz * Iz;
  cvt_f32_bf16<<<(nx + 255) / 256, 256, 0, stream>>>(x, xb, nx);
  cvt_f32_bf16<<<(Gz * Iz + 255) / 256, 256, 0, stream>>>(W_ih, wihb, Gz * Iz);
  cvt_f32_bf16<<<(Gz * Hz + 255) / 256, 256, 0, stream>>>(W_hh, whhb, Gz * Hz);
  cvt_f32_bf16<<<(Oz * Hz + 255) / 256, 256, 0, stream>>>(W_fc, wfcb, Oz * Hz);
  bias_sum<<<(Gz + 255) / 256, 256, 0, stream>>>(b_ih, b_hh, biasg);

  // Phase 1: input projection GEMM (WMMA bf16)
  xproj_gemm<<<(Bz * Tz / 128) * (Gz / 32), 256, 0, stream>>>(xb, wihb, biasg, xp);
  // Phase 2: serial recurrence, batch-split across 2 persistent blocks,
  //          x_proj double-buffered in LDS via async-to-LDS loads
  lstm_scan<<<2, 1024, 0, stream>>>(whhb, xp, hsb);
  // Phase 3: FC + softmax (WMMA bf16 + shfl reductions)
  fc_softmax<<<Bz * Tz / 128, 256, 0, stream>>>(hsb, wfcb, b_fc, out);
}
